// GATLayer_58042188038756
// MI455X (gfx1250) — matrix operand order repaired and checked by reading, not compiled, not run on hardware
//
#include <hip/hip_runtime.h>
#include <hip/hip_bf16.h>
#include <stdint.h>

#define N_NODES 8192
#define IN_DIM  128
#define OUT_DIM 64
#define LRELU_ALPHA 0.2f
#define NEG_BIG -9.0e15f

typedef __attribute__((ext_vector_type(16))) __bf16        v16bf;
typedef __attribute__((ext_vector_type(8)))  float         v8f;
typedef __attribute__((ext_vector_type(4)))  unsigned int  u32x4;
typedef __attribute__((ext_vector_type(4)))  float         f32x4;

__device__ __forceinline__ unsigned short f2bf(float f) {
  unsigned u = __float_as_uint(f);
  unsigned r = ((u >> 16) & 1u) + 0x7fffu;   // round-to-nearest-even
  return (unsigned short)((u + r) >> 16);
}

// ---------------------------------------------------------------------------
// Kernel 1: h = x @ W (f32 exact). Writes hT in *tile-major* bf16 layout:
//   hTt[(i>>5)*2048 + n*32 + (i&31)] = bf16(h[i][n])
// so the attention kernel's WMMA B-fragment (lane ml, K=16*half..+15 of a
// 32-neighbor chunk, column n0+ml) is one contiguous 32-byte load.
// Also reduces s = h.a_src, d = h.a_dst per row.
// ---------------------------------------------------------------------------
__global__ __launch_bounds__(256) void k_hproj(
    const float* __restrict__ x, const float* __restrict__ W,
    const float* __restrict__ a,
    unsigned short* __restrict__ hTt, float* __restrict__ sv, float* __restrict__ dvec)
{
  __shared__ float Wl[IN_DIM * OUT_DIM];   // 32 KB
  const int t = threadIdx.x, lane = t & 31, wave = t >> 5;
  for (int idx = t; idx < IN_DIM * OUT_DIM; idx += 256) Wl[idx] = W[idx];
  __syncthreads();

  const float as0 = a[lane],      as1 = a[lane + 32];
  const float ad0 = a[64 + lane], ad1 = a[96 + lane];
  const int i0 = blockIdx.x * 64 + wave * 8;

  for (int rr = 0; rr < 8; ++rr) {
    const int i = i0 + rr;
    const float* xr = x + (size_t)i * IN_DIM;
    float xrg[4] = { xr[lane], xr[lane + 32], xr[lane + 64], xr[lane + 96] };
    float h0 = 0.f, h1 = 0.f;
#pragma unroll
    for (int kk = 0; kk < 4; ++kk) {
      for (int k = 0; k < 32; ++k) {
        const float xa = __shfl(xrg[kk], k, 32);     // wave32 broadcast
        const int kg = kk * 32 + k;
        h0 = fmaf(xa, Wl[kg * 64 + lane],      h0);
        h1 = fmaf(xa, Wl[kg * 64 + lane + 32], h1);
      }
    }
    const size_t tb = (size_t)(i >> 5) * 2048 + (i & 31);
    hTt[tb + (size_t)lane        * 32] = f2bf(h0);
    hTt[tb + (size_t)(lane + 32) * 32] = f2bf(h1);
    float ps = h0 * as0 + h1 * as1;
    float pd = h0 * ad0 + h1 * ad1;
#pragma unroll
    for (int off = 16; off; off >>= 1) {
      ps += __shfl_xor(ps, off, 32);
      pd += __shfl_xor(pd, off, 32);
    }
    if (lane == 0) { sv[i] = ps; dvec[i] = pd; }
  }
}

// ---------------------------------------------------------------------------
// Kernel 2: fused masked-softmax attention, flash style, WMMA bf16.
// No LDS, no barriers: waves fully independent; hTt/dvec are L2-resident.
// adj (the one-pass 256MB HBM stream) is software-pipelined one chunk ahead
// with non-temporal b128 loads.
// ---------------------------------------------------------------------------
__global__ __launch_bounds__(128) void k_attn(
    const int* __restrict__ adj, const unsigned short* __restrict__ hTt,
    const float* __restrict__ sv, const float* __restrict__ dvec,
    float* __restrict__ m_p, float* __restrict__ l_p, float* __restrict__ acc_p,
    int jlen)
{
  const int t = threadIdx.x, lane = t & 31, wave = t >> 5;
  const int half = lane >> 4, ml = lane & 15;
  const int rowbase = blockIdx.x * 64 + wave * 16;
  const int i = rowbase + ml;                       // this lane's attention row
  const int by = blockIdx.y;
  const int jbase = by * jlen;
  const int jend = jbase + jlen;

  const float s_i = sv[i];
  const int* arow = adj + (size_t)i * N_NODES;

  float mrow = -__builtin_inff();
  float lrow = 0.f;
  v8f acc0 = {}, acc1 = {}, acc2 = {}, acc3 = {};

  // ---- prefetch first adj chunk (A layout: e -> K = 16*half+e)
  int cpf = jbase + 16 * half;
  u32x4 a0 = __builtin_nontemporal_load((const u32x4*)(arow + cpf));
  u32x4 a1 = __builtin_nontemporal_load((const u32x4*)(arow + cpf + 4));
  u32x4 a2 = __builtin_nontemporal_load((const u32x4*)(arow + cpf + 8));
  u32x4 a3 = __builtin_nontemporal_load((const u32x4*)(arow + cpf + 12));

  for (int j0 = jbase; j0 < jend; j0 += 32) {
    const u32x4 c0v = a0, c1v = a1, c2v = a2, c3v = a3;

    // ---- prefetch next chunk's adj (wrap to a valid address on last iter)
    const int jn = (j0 + 32 < jend) ? (j0 + 32) : jbase;
    cpf = jn + 16 * half;
    a0 = __builtin_nontemporal_load((const u32x4*)(arow + cpf));
    a1 = __builtin_nontemporal_load((const u32x4*)(arow + cpf + 4));
    a2 = __builtin_nontemporal_load((const u32x4*)(arow + cpf + 8));
    a3 = __builtin_nontemporal_load((const u32x4*)(arow + cpf + 12));

    // ---- d values for this lane's A columns (L2-resident broadcast)
    const float* dp = dvec + j0;
    f32x4 d0 = *(const f32x4*)(dp + 16 * half);
    f32x4 d1 = *(const f32x4*)(dp + 16 * half + 4);
    f32x4 d2 = *(const f32x4*)(dp + 16 * half + 8);
    f32x4 d3 = *(const f32x4*)(dp + 16 * half + 12);

    unsigned avv[16]; float dvv[16];
#pragma unroll
    for (int c = 0; c < 4; ++c) {
      avv[c] = c0v[c]; avv[4 + c] = c1v[c]; avv[8 + c] = c2v[c]; avv[12 + c] = c3v[c];
      dvv[c] = d0[c];  dvv[4 + c] = d1[c];  dvv[8 + c] = d2[c];  dvv[12 + c] = d3[c];
    }

    // ---- e = mask(leakyrelu(s_i + d_j)), online-softmax update ----
    float ev[16];
    float mloc = -__builtin_inff();
#pragma unroll
    for (int e = 0; e < 16; ++e) {
      float v = s_i + dvv[e];
      v = v > 0.f ? v : LRELU_ALPHA * v;
      v = avv[e] ? v : NEG_BIG;
      ev[e] = v;
      mloc = fmaxf(mloc, v);
    }
    mloc = fmaxf(mloc, __shfl_xor(mloc, 16, 32));   // combine K-halves of row
    const float mnew = fmaxf(mrow, mloc);
    const float scale = __expf(mrow - mnew);        // exp(-inf)=0 first pass

    float rs = 0.f;
    v16bf Av;
#pragma unroll
    for (int e = 0; e < 16; ++e) {
      float p = __expf(ev[e] - mnew);
      rs += p;
      Av[e] = (__bf16)p;
    }
    rs += __shfl_xor(rs, 16, 32);
    lrow = lrow * scale + rs;
    mrow = mnew;

    // ---- rescale accumulators: C/D layout row M = r + 8*half ----
#pragma unroll
    for (int r = 0; r < 8; ++r) {
      const float sc = __shfl(scale, r + 8 * half, 32);
      acc0[r] *= sc; acc1[r] *= sc; acc2[r] *= sc; acc3[r] *= sc;
    }

    // ---- B fragments: contiguous 32B per lane from tile-major hTt ----
    const unsigned short* hb = hTt + (size_t)(j0 >> 5) * 2048 + 16 * half;
    const v16bf B0 = *(const v16bf*)(hb + ( 0 + ml) * 32);
    const v16bf B1 = *(const v16bf*)(hb + (16 + ml) * 32);
    const v16bf B2 = *(const v16bf*)(hb + (32 + ml) * 32);
    const v16bf B3 = *(const v16bf*)(hb + (48 + ml) * 32);

    acc0 = __builtin_amdgcn_wmma_f32_16x16x32_bf16(false, Av, false, B0, (short)0, acc0, false, false);
    acc1 = __builtin_amdgcn_wmma_f32_16x16x32_bf16(false, Av, false, B1, (short)0, acc1, false, false);
    acc2 = __builtin_amdgcn_wmma_f32_16x16x32_bf16(false, Av, false, B2, (short)0, acc2, false, false);
    acc3 = __builtin_amdgcn_wmma_f32_16x16x32_bf16(false, Av, false, B3, (short)0, acc3, false, false);
  }

  // ---- write split-partials ----
  float* accout = acc_p + (size_t)by * N_NODES * 64;
#pragma unroll
  for (int r = 0; r < 8; ++r) {
    const int row = rowbase + r + 8 * half;
    accout[(size_t)row * 64 +  0 + ml] = acc0[r];
    accout[(size_t)row * 64 + 16 + ml] = acc1[r];
    accout[(size_t)row * 64 + 32 + ml] = acc2[r];
    accout[(size_t)row * 64 + 48 + ml] = acc3[r];
  }
  if (half == 0) {
    m_p[(size_t)by * N_NODES + rowbase + ml] = mrow;
    l_p[(size_t)by * N_NODES + rowbase + ml] = lrow;
  }
}

// ---------------------------------------------------------------------------
// Kernel 3: merge split partials (softmax renorm), divide, ELU.
// ---------------------------------------------------------------------------
__global__ __launch_bounds__(256) void k_combine(
    const float* __restrict__ m_p, const float* __restrict__ l_p,
    const float* __restrict__ acc_p, float* __restrict__ out, int nsplit)
{
  const int idx = blockIdx.x * 256 + threadIdx.x;
  const int i = idx >> 6, n = idx & 63;
  float M = -__builtin_inff();
  for (int s = 0; s < nsplit; ++s) M = fmaxf(M, m_p[(size_t)s * N_NODES + i]);
  float L = 0.f, A = 0.f;
  for (int s = 0; s < nsplit; ++s) {
    const float w = __expf(m_p[(size_t)s * N_NODES + i] - M);
    L += l_p[(size_t)s * N_NODES + i] * w;
    A += acc_p[((size_t)s * N_NODES + i) * 64 + n] * w;
  }
  const float o = A / L;
  out[idx] = o > 0.f ? o : (__expf(o) - 1.f);
}

// ---------------------------------------------------------------------------
extern "C" void kernel_launch(void* const* d_in, const int* in_sizes, int n_in,
                              void* d_out, int out_size, void* d_ws, size_t ws_size,
                              hipStream_t stream)
{
  (void)in_sizes; (void)n_in; (void)out_size;
  const float* x   = (const float*)d_in[0];
  const int*   adj = (const int*)d_in[1];
  const float* W   = (const float*)d_in[2];
  const float* a   = (const float*)d_in[3];
  float* out = (float*)d_out;

  char* ws = (char*)d_ws;
  size_t off = 0;
  auto alloc = [&](size_t bytes) -> char* {
    char* p = ws + off;
    off = (off + bytes + 255) & ~(size_t)255;
    return p;
  };

  unsigned short* hTt = (unsigned short*)alloc((size_t)OUT_DIM * N_NODES * 2);  // 1 MB
  float* sv   = (float*)alloc((size_t)N_NODES * 4);
  float* dvec = (float*)alloc((size_t)N_NODES * 4);

  int split = 8;   // split the neighbor axis for occupancy; shrink if ws is small
  while (split > 1) {
    size_t need = off + (size_t)split * N_NODES * 4 * 2
                      + (size_t)split * N_NODES * 64 * 4 + 1024;
    if (need <= ws_size) break;
    split >>= 1;
  }
  float* m_p   = (float*)alloc((size_t)split * N_NODES * 4);
  float* l_p   = (float*)alloc((size_t)split * N_NODES * 4);
  float* acc_p = (float*)alloc((size_t)split * N_NODES * 64 * 4);

  k_hproj<<<N_NODES / 64, 256, 0, stream>>>(x, W, a, hTt, sv, dvec);
  dim3 g2(N_NODES / 64, split);
  k_attn<<<g2, 128, 0, stream>>>(adj, hTt, sv, dvec, m_p, l_p, acc_p, N_NODES / split);
  k_combine<<<(N_NODES * 64) / 256, 256, 0, stream>>>(m_p, l_p, acc_p, out, split);
}
